// TreeLstm_88519275971244
// MI455X (gfx1250) — compile-verified, matrix operand order checked by reading, not run on hardware
//
#include <hip/hip_runtime.h>

typedef __attribute__((ext_vector_type(16))) _Float16 v16h;
typedef __attribute__((ext_vector_type(8)))  _Float16 v8h;
typedef __attribute__((ext_vector_type(8)))  float    v8f;

namespace {
constexpr int kH    = 256;          // hidden size
constexpr int kGN   = 5 * kH;       // 1280 gate columns (i, fl, fr, o, u)
constexpr int kD    = 300;          // embedding dim
constexpr int kDP   = 320;          // padded to multiple of 32 (WMMA K-tile)
constexpr int kHH   = 2 * kH;       // 512 = internal-node input dim
constexpr int kNL   = 16384;        // number of leaves
constexpr int kRows = 2 * kNL - 1;  // 32767 total tree nodes
}

// ---------------- prep kernels: f32 -> f16, transpose, pad, gather ----------

__global__ __launch_bounds__(256) void prep_wx(const float* __restrict__ Wx,
                                               _Float16* __restrict__ WxT) {
  int idx = blockIdx.x * 256 + threadIdx.x;
  if (idx >= kGN * kDP) return;
  int n = idx / kDP, k = idx - n * kDP;          // WxT[n][k] = Wx[k][n], pad k>=300
  float v = (k < kD) ? Wx[(size_t)k * kGN + n] : 0.f;
  WxT[idx] = (_Float16)v;
}

__global__ __launch_bounds__(256) void prep_u(const float* __restrict__ U,
                                              _Float16* __restrict__ UT) {
  int idx = blockIdx.x * 256 + threadIdx.x;
  if (idx >= kGN * kHH) return;
  int n = idx / kHH, k = idx - n * kHH;          // UT[n][k] = U[k][n]
  UT[idx] = (_Float16)U[(size_t)k * kGN + n];
}

__global__ __launch_bounds__(256) void gather_leaves(const int* __restrict__ ids,
                                                     const float* __restrict__ emb,
                                                     _Float16* __restrict__ X) {
  int idx = blockIdx.x * 256 + threadIdx.x;
  if (idx >= kNL * kDP) return;
  int m = idx / kDP, k = idx - m * kDP;
  float v = (k < kD) ? emb[(size_t)ids[m] * kD + k] : 0.f;
  X[idx] = (_Float16)v;
}

// ---------------- shared helpers -------------------------------------------

__device__ __forceinline__ float sigm(float x) { return 1.f / (1.f + __expf(-x)); }

// ---- hand-clause B-fragment loads ------------------------------------------
// A wave's 10 tiles sit at constant byte offsets (g*256 + s*16)*K*2 from one
// base: issue them as one clause per asm block so ALL loads are in flight
// before a single s_wait_loadcnt 0x0, instead of load->wait0->wmma ten times.

// tiles t0..t4 (g0s0,g0s1,g1s0,g1s1,g2s0) -> bf[0..9]
template<int K>
__device__ __forceinline__ void load_b_lo(uint64_t base, v8h* bf) {
  constexpr int GB = 256 * K * 2;   // gate-block byte stride
  constexpr int SB = 16 * K * 2;    // sub-tile byte stride
  asm volatile(
      "s_clause 0x9\n\t"
      "global_load_b128 %0, %10, off\n\t"
      "global_load_b128 %1, %10, off offset:32\n\t"
      "global_load_b128 %2, %10, off offset:%c11\n\t"
      "global_load_b128 %3, %10, off offset:%c12\n\t"
      "global_load_b128 %4, %10, off offset:%c13\n\t"
      "global_load_b128 %5, %10, off offset:%c14\n\t"
      "global_load_b128 %6, %10, off offset:%c15\n\t"
      "global_load_b128 %7, %10, off offset:%c16\n\t"
      "global_load_b128 %8, %10, off offset:%c17\n\t"
      "global_load_b128 %9, %10, off offset:%c18\n\t"
      : "=v"(bf[0]), "=v"(bf[1]), "=v"(bf[2]), "=v"(bf[3]), "=v"(bf[4]),
        "=v"(bf[5]), "=v"(bf[6]), "=v"(bf[7]), "=v"(bf[8]), "=v"(bf[9])
      : "v"(base),
        "n"(SB), "n"(SB + 32), "n"(GB), "n"(GB + 32),
        "n"(GB + SB), "n"(GB + SB + 32), "n"(2 * GB), "n"(2 * GB + 32)
      : "memory");
}

// tiles t5..t9 (g2s1,g3s0,g3s1,g4s0,g4s1) -> bf[0..9]
template<int K>
__device__ __forceinline__ void load_b_hi(uint64_t base, v8h* bf) {
  constexpr int GB = 256 * K * 2;
  constexpr int SB = 16 * K * 2;
  asm volatile(
      "s_clause 0x9\n\t"
      "global_load_b128 %0, %10, off offset:%c11\n\t"
      "global_load_b128 %1, %10, off offset:%c12\n\t"
      "global_load_b128 %2, %10, off offset:%c13\n\t"
      "global_load_b128 %3, %10, off offset:%c14\n\t"
      "global_load_b128 %4, %10, off offset:%c15\n\t"
      "global_load_b128 %5, %10, off offset:%c16\n\t"
      "global_load_b128 %6, %10, off offset:%c17\n\t"
      "global_load_b128 %7, %10, off offset:%c18\n\t"
      "global_load_b128 %8, %10, off offset:%c19\n\t"
      "global_load_b128 %9, %10, off offset:%c20\n\t"
      : "=v"(bf[0]), "=v"(bf[1]), "=v"(bf[2]), "=v"(bf[3]), "=v"(bf[4]),
        "=v"(bf[5]), "=v"(bf[6]), "=v"(bf[7]), "=v"(bf[8]), "=v"(bf[9])
      : "v"(base),
        "n"(2 * GB + SB), "n"(2 * GB + SB + 32), "n"(3 * GB), "n"(3 * GB + 32),
        "n"(3 * GB + SB), "n"(3 * GB + SB + 32), "n"(4 * GB), "n"(4 * GB + 32),
        "n"(4 * GB + SB), "n"(4 * GB + SB + 32)
      : "memory");
}

// tail: tiles g0..g4 at stride GB only -> bf[0..9]
template<int K>
__device__ __forceinline__ void load_b_tail(uint64_t base, v8h* bf) {
  constexpr int GB = 256 * K * 2;
  asm volatile(
      "s_clause 0x9\n\t"
      "global_load_b128 %0, %10, off\n\t"
      "global_load_b128 %1, %10, off offset:32\n\t"
      "global_load_b128 %2, %10, off offset:%c11\n\t"
      "global_load_b128 %3, %10, off offset:%c12\n\t"
      "global_load_b128 %4, %10, off offset:%c13\n\t"
      "global_load_b128 %5, %10, off offset:%c14\n\t"
      "global_load_b128 %6, %10, off offset:%c15\n\t"
      "global_load_b128 %7, %10, off offset:%c16\n\t"
      "global_load_b128 %8, %10, off offset:%c17\n\t"
      "global_load_b128 %9, %10, off offset:%c18\n\t"
      : "=v"(bf[0]), "=v"(bf[1]), "=v"(bf[2]), "=v"(bf[3]), "=v"(bf[4]),
        "=v"(bf[5]), "=v"(bf[6]), "=v"(bf[7]), "=v"(bf[8]), "=v"(bf[9])
      : "v"(base),
        "n"(GB), "n"(GB + 32), "n"(2 * GB), "n"(2 * GB + 32),
        "n"(3 * GB), "n"(3 * GB + 32), "n"(4 * GB), "n"(4 * GB + 32)
      : "memory");
}

// single wait; tied "+v" pass-throughs pin every WMMA after this instruction
// and force all fragment live ranges to overlap (distinct registers).
__device__ __forceinline__ void wait_loads20(v8h* bf) {
  asm volatile("s_wait_loadcnt 0x0"
      : "+v"(bf[0]),  "+v"(bf[1]),  "+v"(bf[2]),  "+v"(bf[3]),  "+v"(bf[4]),
        "+v"(bf[5]),  "+v"(bf[6]),  "+v"(bf[7]),  "+v"(bf[8]),  "+v"(bf[9]),
        "+v"(bf[10]), "+v"(bf[11]), "+v"(bf[12]), "+v"(bf[13]), "+v"(bf[14]),
        "+v"(bf[15]), "+v"(bf[16]), "+v"(bf[17]), "+v"(bf[18]), "+v"(bf[19])
      :: "memory");
}

__device__ __forceinline__ void wait_loads10(v8h* bf) {
  asm volatile("s_wait_loadcnt 0x0"
      : "+v"(bf[0]), "+v"(bf[1]), "+v"(bf[2]), "+v"(bf[3]), "+v"(bf[4]),
        "+v"(bf[5]), "+v"(bf[6]), "+v"(bf[7]), "+v"(bf[8]), "+v"(bf[9])
      :: "memory");
}

__device__ __forceinline__ v16h mk_frag(v8h lo, v8h hi) {
  return __builtin_shufflevector(lo, hi,
      0, 1, 2, 3, 4, 5, 6, 7, 8, 9, 10, 11, 12, 13, 14, 15);
}

// ---------------- fused level GEMM + LSTM cell ------------------------------
// C_gates[16 x 1280] = A[16 x K] @ Wt^T, then per-column LSTM epilogue.
// 8 waves / block; wave w owns gate-columns [w*32, w*32+32) of every gate
// block (tiles g*16 + w*2 + {0,1}), so all 5 gate values for one output
// element live in this wave's accumulators -> in-register epilogue.
// A tiles are double-buffered in LDS via async global->LDS DMA (ASYNCcnt).

template<int K>
__global__ __launch_bounds__(256)
void tree_level(const _Float16* __restrict__ A,    // [n][K] f16 (lda == K)
                const _Float16* __restrict__ Wt,   // [kGN][K] f16, pre-transposed
                const float* __restrict__ bias,    // [kGN]
                const float* __restrict__ cprev,   // [2n][kH] f32, null for leaves
                float* __restrict__ cout,          // [n][kH] f32
                _Float16* __restrict__ hout,       // [n][kH] f16
                int n) {
  __shared__ __align__(16) _Float16 Atile[2][16 * 32];   // 2 x 1KB K-slabs of A
  const int tid  = threadIdx.x;
  const int wave = tid >> 5;
  const int lane = tid & 31;
  const int half = lane >> 4;     // WMMA half-wave select
  const int l16  = lane & 15;
  const int m0   = blockIdx.x * 16;

  v8f acc[5][2];
  #pragma unroll
  for (int g = 0; g < 5; ++g)
    #pragma unroll
    for (int s = 0; s < 2; ++s)
      #pragma unroll
      for (int e = 0; e < 8; ++e) acc[g][s][e] = 0.f;

  constexpr int nkt = K / 32;

  // async stage of one 16x32 f16 tile: 64 lanes (waves 0-1) x 16B
  const int srow = tid >> 2;   // 0..15
  const int schk = tid & 3;    // 16B chunk within 64B row
  auto stage = [&](int kt, int buf) {
    if (tid < 64) {
      uint64_t g = (uint64_t)(uintptr_t)A +
                   2ull * ((size_t)(m0 + srow) * K + kt * 32) + schk * 16;
      uint32_t l = (uint32_t)(uintptr_t)(&Atile[buf][0]) + srow * 64 + schk * 16;
      asm volatile("global_load_async_to_lds_b128 %0, %1, off"
                   :: "v"(l), "v"(g) : "memory");
    }
  };

  stage(0, 0);
  for (int kt = 0; kt < nkt; ++kt) {
    const bool pf = (kt + 1 < nkt);
    if (pf) {
      stage(kt + 1, (kt + 1) & 1);
      asm volatile("s_wait_asynccnt 0x1" ::: "memory");  // tile kt done (in-order)
    } else {
      asm volatile("s_wait_asynccnt 0x0" ::: "memory");
    }

    // two clauses of 10 B loads, then ONE load wait for the whole K-step
    v8h bf[20];
    uint64_t wbase = (uint64_t)(uintptr_t)Wt +
                     2ull * ((size_t)(wave * 32 + l16) * K + kt * 32 + half * 8);
    load_b_lo<K>(wbase, bf);
    load_b_hi<K>(wbase, bf + 10);

    __syncthreads();   // publish A tile kt to all waves

    // A fragment per ISA 16-bit A layout:
    // lanes 0-15: row=l16, K 0-7 / K 16-23; lanes 16-31: K 8-15 / K 24-31
    const v8h* At = (const v8h*)Atile[kt & 1];
    v8h a_lo = At[l16 * 4 + half];
    v8h a_hi = At[l16 * 4 + 2 + half];
    v16h afrag = mk_frag(a_lo, a_hi);

    wait_loads20(bf);

    #pragma unroll
    for (int g = 0; g < 5; ++g)
      #pragma unroll
      for (int s = 0; s < 2; ++s) {
        int t = g * 2 + s;
        acc[g][s] = __builtin_amdgcn_wmma_f32_16x16x32_f16(
            false, afrag, false, mk_frag(bf[2 * t], bf[2 * t + 1]),
            (short)0, acc[g][s], false, false);
      }

    __syncthreads();   // everyone done reading buf before it is refilled
  }

  // ---- LSTM cell epilogue, fully in-register ----
  #pragma unroll
  for (int s = 0; s < 2; ++s) {
    const int col = wave * 32 + s * 16 + l16;   // 0..255 inside each gate block
    #pragma unroll
    for (int r = 0; r < 8; ++r) {
      const int row = m0 + r + half * 8;        // C/D layout: VGPR r, +8 per half
      if (row < n) {
        float gi = acc[0][s][r] + bias[0 * kH + col];
        float gl = acc[1][s][r] + bias[1 * kH + col];
        float gr = acc[2][s][r] + bias[2 * kH + col];
        float go = acc[3][s][r] + bias[3 * kH + col];
        float gu = acc[4][s][r] + bias[4 * kH + col];
        float c = sigm(gi) * tanhf(gu);
        if (cprev) {
          c += sigm(gl) * cprev[(size_t)(2 * row)     * kH + col] +
               sigm(gr) * cprev[(size_t)(2 * row + 1) * kH + col];
        }
        cout[(size_t)row * kH + col] = c;
        hout[(size_t)row * kH + col] = (_Float16)(sigm(go) * tanhf(c));
      }
    }
  }
}

// ---------------- fused tail: levels nn = 16,8,4,2,1 ------------------------
// One 512-thread workgroup = 16 waves (4 per SIMD for latency hiding); wave w
// owns gate-columns [w*16, w*16+16) of every gate block (5 tiles, one per
// gate). h (f16) and c (f32) stay resident in LDS across levels (96KB of the
// WGP's 320KB); A fragments come straight from the LDS h buffer; h rows are
// also streamed to Hall for the final prediction GEMM.

__global__ __launch_bounds__(512)
void tree_tail(const _Float16* __restrict__ hin,   // [32][kH] f16 (nn=32 level h)
               const float* __restrict__ cin,      // [32][kH] f32
               const _Float16* __restrict__ UT,    // [kGN][kHH] f16
               const float* __restrict__ bias,
               _Float16* __restrict__ HallTail) {  // Hall + tail offset
  extern __shared__ __align__(16) char smem[];
  _Float16* hbuf[2];
  float*    cbuf[2];
  hbuf[0] = (_Float16*)smem;                // 16KB
  hbuf[1] = hbuf[0] + 32 * kH;              // 16KB
  cbuf[0] = (float*)(hbuf[1] + 32 * kH);    // 32KB
  cbuf[1] = cbuf[0] + 32 * kH;              // 32KB

  const int tid  = threadIdx.x;
  const int wave = tid >> 5;      // 0..15
  const int lane = tid & 31;
  const int half = lane >> 4;
  const int l16  = lane & 15;

  for (int i = tid; i < 32 * kH; i += 512) {
    hbuf[0][i] = hin[i];
    cbuf[0][i] = cin[i];
  }
  __syncthreads();

  int in = 0;
  size_t off = 0;
  for (int nn = 16; nn >= 1; nn >>= 1) {
    const int out = in ^ 1;
    // zero output h so pad rows feed zeros into the next level's A tile
    for (int i = tid; i < 32 * (kH / 2); i += 512) ((uint32_t*)hbuf[out])[i] = 0u;
    __syncthreads();

    v8f acc[5];
    #pragma unroll
    for (int g = 0; g < 5; ++g)
      #pragma unroll
      for (int e = 0; e < 8; ++e) acc[g][e] = 0.f;

    const v8h* At = (const v8h*)hbuf[in];   // [16][512] f16 view, 64 v8h per row
    for (int kt = 0; kt < kHH / 32; ++kt) {
      v8h bf[10];
      uint64_t wbase = (uint64_t)(uintptr_t)UT +
                       2ull * ((size_t)(wave * 16 + l16) * kHH + kt * 32 + half * 8);
      load_b_tail<kHH>(wbase, bf);

      v8h a_lo = At[l16 * 64 + kt * 4 + half];
      v8h a_hi = At[l16 * 64 + kt * 4 + 2 + half];
      v16h afrag = mk_frag(a_lo, a_hi);

      wait_loads10(bf);

      #pragma unroll
      for (int g = 0; g < 5; ++g)
        acc[g] = __builtin_amdgcn_wmma_f32_16x16x32_f16(
            false, afrag, false, mk_frag(bf[2 * g], bf[2 * g + 1]),
            (short)0, acc[g], false, false);
    }

    const int col = wave * 16 + l16;        // 0..255 inside each gate block
    #pragma unroll
    for (int r = 0; r < 8; ++r) {
      const int row = r + half * 8;
      if (row < nn) {
        float gi = acc[0][r] + bias[0 * kH + col];
        float gl = acc[1][r] + bias[1 * kH + col];
        float gr = acc[2][r] + bias[2 * kH + col];
        float go = acc[3][r] + bias[3 * kH + col];
        float gu = acc[4][r] + bias[4 * kH + col];
        float c = sigm(gi) * tanhf(gu) +
                  sigm(gl) * cbuf[in][(2 * row)     * kH + col] +
                  sigm(gr) * cbuf[in][(2 * row + 1) * kH + col];
        cbuf[out][row * kH + col] = c;
        _Float16 h = (_Float16)(sigm(go) * tanhf(c));
        hbuf[out][row * kH + col] = h;
        HallTail[off + (size_t)row * kH + col] = h;
      }
    }
    __syncthreads();
    off += (size_t)nn * kH;
    in = out;
  }
}

// ---------------- prediction: all_h (32767x256) @ Wp (256x5) + bp ----------

__global__ __launch_bounds__(256)
void predict(const _Float16* __restrict__ Hall, const float* __restrict__ Wp,
             const float* __restrict__ bp, float* __restrict__ out, int nrows) {
  __shared__ float wps[kH * 5];
  for (int i = threadIdx.x; i < kH * 5; i += 256) wps[i] = Wp[i];
  __syncthreads();
  int rid = blockIdx.x * 256 + threadIdx.x;
  if (rid >= nrows) return;
  const v8h* hrow = (const v8h*)(Hall + (size_t)rid * kH);
  float s0 = bp[0], s1 = bp[1], s2 = bp[2], s3 = bp[3], s4 = bp[4];
  for (int j8 = 0; j8 < kH / 8; ++j8) {
    v8h hv = hrow[j8];
    #pragma unroll
    for (int e = 0; e < 8; ++e) {
      float h = (float)hv[e];
      int j = j8 * 8 + e;
      s0 += h * wps[j * 5 + 0];
      s1 += h * wps[j * 5 + 1];
      s2 += h * wps[j * 5 + 2];
      s3 += h * wps[j * 5 + 3];
      s4 += h * wps[j * 5 + 4];
    }
  }
  float* o = out + (size_t)rid * 5;
  o[0] = s0; o[1] = s1; o[2] = s2; o[3] = s3; o[4] = s4;
}

// ---------------- host launcher --------------------------------------------

extern "C" void kernel_launch(void* const* d_in, const int* in_sizes, int n_in,
                              void* d_out, int out_size, void* d_ws, size_t ws_size,
                              hipStream_t stream) {
  (void)in_sizes; (void)n_in; (void)out_size; (void)ws_size;
  const int*   ids = (const int*)  d_in[0];
  const float* emb = (const float*)d_in[1];
  const float* Wx  = (const float*)d_in[2];
  const float* U   = (const float*)d_in[3];
  const float* b   = (const float*)d_in[4];
  const float* Wp  = (const float*)d_in[5];
  const float* bpv = (const float*)d_in[6];

  // workspace carve (256B aligned): ~55 MB total
  char* p = (char*)d_ws;
  auto carve = [&](size_t bytes) {
    char* q = p; p += (bytes + 255) & ~(size_t)255; return q;
  };
  _Float16* WxT  = (_Float16*)carve((size_t)kGN * kDP * 2);          // 0.82 MB
  _Float16* UT   = (_Float16*)carve((size_t)kGN * kHH * 2);          // 1.31 MB
  _Float16* X16  = (_Float16*)carve((size_t)kNL * kDP * 2);          // 10.5 MB
  // all h levels, f16, level-contiguous; +16 rows of [512] pad for OOB A reads
  _Float16* Hall = (_Float16*)carve(((size_t)kRows * kH + 16 * kHH) * 2);
  float* cbuf0   = (float*)carve((size_t)kNL * kH * 4);              // 16.8 MB
  float* cbuf1   = (float*)carve((size_t)(kNL / 2) * kH * 4);        // 8.4 MB

  prep_wx<<<(kGN * kDP + 255) / 256, 256, 0, stream>>>(Wx, WxT);
  prep_u <<<(kGN * kHH + 255) / 256, 256, 0, stream>>>(U, UT);
  gather_leaves<<<(kNL * kDP + 255) / 256, 256, 0, stream>>>(ids, emb, X16);

  // leaf level: c = i*u (children are zero)
  tree_level<kDP><<<kNL / 16, 256, 0, stream>>>(X16, WxT, b,
                                                nullptr, cbuf0, Hall, kNL);

  // internal levels nn = 8192 .. 32:
  // hh[j] = [h[2j], h[2j+1]] == previous h buffer viewed [n/2][512]
  size_t off = (size_t)kNL * kH;
  const _Float16* hprev = Hall;
  float* cp = cbuf0;
  float* cn = cbuf1;
  int nrows = kNL;
  for (int lvl = 1; lvl <= 9; ++lvl) {
    int nn = nrows >> 1;
    _Float16* hcur = Hall + off;
    tree_level<kHH><<<(nn + 15) / 16, 256, 0, stream>>>(hprev, UT, b,
                                                        cp, cn, hcur, nn);
    off += (size_t)nn * kH;
    hprev = hcur;
    float* t = cp; cp = cn; cn = t;
    nrows = nn;
  }

  // fused tail: levels nn = 16,8,4,2,1 in one 512-thread workgroup
  tree_tail<<<1, 512, 96 * 1024, stream>>>(hprev, cp, UT, b, Hall + off);

  predict<<<(kRows + 255) / 256, 256, 0, stream>>>(Hall, Wp, bpv,
                                                   (float*)d_out, kRows);
}
